// TtoM_70720931496569
// MI455X (gfx1250) — matrix operand order repaired and checked by reading, not compiled, not run on hardware
//
#include <hip/hip_runtime.h>

typedef __attribute__((ext_vector_type(16))) _Float16 v16h;
typedef __attribute__((ext_vector_type(8)))  _Float16 v8h;
typedef __attribute__((ext_vector_type(8)))  float    v8f;

#define BATCH 512
#define CDIM  24      // channels == head dim DK
#define HW    3136    // 56*56 queries
#define MTOK  49      // kv tokens
#define DDIM  768     // z feature dim
#define QTILES 196    // 3136/16 query tiles
#define KV_STRIDE 2048            // per-batch padded: K[64][32] / Vt[32][64] halves
#define SCALE 0.20412414523193154f // 24^-0.5

static __device__ __forceinline__ v8f wmma16(v16h a, v16h b, v8f c) {
  // D(f32 16x16) = A(f16 16x32) * B(f16 32x16) + C
  return __builtin_amdgcn_wmma_f32_16x16x32_f16(
      /*neg_a=*/false, a, /*neg_b=*/false, b,
      /*c_mod=*/(short)0, c, /*reuse_a=*/false, /*reuse_b=*/false);
}

// ---------------------------------------------------------------------------
// Kernel 0: pre-transpose + f16-convert the tiny projection weights:
// Wt[m][n][d] = W_m[d][n],  m in {K,V}, n<24, d<768.  (2*18432 halves)
// Makes kernel 1's B fragments contiguous f16 -> 2x b128 loads, no cvt.
// ---------------------------------------------------------------------------
__global__ __launch_bounds__(256) void wt_kernel(
    const float* __restrict__ Wk, const float* __restrict__ Wv,
    _Float16* __restrict__ wt)
{
  const int idx = blockIdx.x * 256 + threadIdx.x;   // 0 .. 2*768*24-1
  const int m   = idx / (DDIM * CDIM);
  const int rem = idx - m * (DDIM * CDIM);
  const int d   = rem / CDIM;
  const int n   = rem - d * CDIM;
  const float* W = m ? Wv : Wk;
  wt[(size_t)m * CDIM * DDIM + n * DDIM + d] = (_Float16)W[rem];
}

// ---------------------------------------------------------------------------
// Kernel 1: K = z@Wk+bk -> Kp[b][64][32] (row-major, padded)
//           V = z@Wv+bv -> Vt[b][32][64] (transposed, padded)
// One block per batch, 8 waves: waves 0-3 -> K M-tiles, waves 4-7 -> V.
// Branch-free: clamped addresses, unified store strides; pads hold finite
// garbage that is provably cancelled downstream.
// ---------------------------------------------------------------------------
__global__ __launch_bounds__(256) void kv_proj_kernel(
    const float* __restrict__ z,
    const _Float16* __restrict__ wt,
    const float* __restrict__ bk, const float* __restrict__ bv,
    _Float16* __restrict__ kp, _Float16* __restrict__ vt)
{
  const int b      = blockIdx.x;
  const int w      = threadIdx.x >> 5;
  const int lane   = threadIdx.x & 31;
  const int laneLo = lane & 15;
  const int laneHi = lane >> 4;
  const int aOff   = laneHi * 8;   // A-frag K pattern {0..7,16..23}/{8..15,24..31}
  const int bOff   = laneHi * 8;   // B-frag K pattern matches A: {0..7,16..23}/{8..15,24..31}

  const int  mtile = w & 3;
  const bool doV   = (w >= 4);
  const float* bias = doV ? bv : bk;
  const _Float16* wm = wt + (doV ? (size_t)CDIM * DDIM : 0);

  const int rowA = mtile * 16 + laneLo;
  const int rowC = rowA < (MTOK - 1) ? rowA : (MTOK - 1);   // clamp pad rows
  const float* zrow = z + ((size_t)b * MTOK + rowC) * DDIM;

  const int n1  = 16 + laneLo;                               // 16..31
  const int n1c = n1 < (CDIM - 1) ? n1 : (CDIM - 1);         // clamp pad cols

  const _Float16* w0row = wm + (size_t)laneLo * DDIM;        // B col n=laneLo
  const _Float16* w1row = wm + (size_t)n1c * DDIM;           // B col n1 (clamped)

  v8f acc0 = {}; v8f acc1 = {};
  for (int kk = 0; kk < DDIM; kk += 32) {
    // A: two contiguous 8-float groups -> 4x global_load_b128 + pk cvts
    const float4* za = (const float4*)(zrow + kk + aOff);
    const float4* zb = (const float4*)(zrow + kk + aOff + 16);
    const float4 q0 = za[0], q1 = za[1], q2 = zb[0], q3 = zb[1];
    v16h a;
    a[0]  = (_Float16)q0.x; a[1]  = (_Float16)q0.y;
    a[2]  = (_Float16)q0.z; a[3]  = (_Float16)q0.w;
    a[4]  = (_Float16)q1.x; a[5]  = (_Float16)q1.y;
    a[6]  = (_Float16)q1.z; a[7]  = (_Float16)q1.w;
    a[8]  = (_Float16)q2.x; a[9]  = (_Float16)q2.y;
    a[10] = (_Float16)q2.z; a[11] = (_Float16)q2.w;
    a[12] = (_Float16)q3.x; a[13] = (_Float16)q3.y;
    a[14] = (_Float16)q3.z; a[15] = (_Float16)q3.w;

    // B: pre-transposed f16 weights -> two 8-half runs (k, k+16) per frag
    const v8h* wp0 = (const v8h*)(w0row + kk + bOff);   // 16B aligned
    const v8h* wp1 = (const v8h*)(w1row + kk + bOff);
    const v8h b0a = wp0[0], b0b = wp0[2];
    const v8h b1a = wp1[0], b1b = wp1[2];
    v16h b0, b1;
    #pragma unroll
    for (int j = 0; j < 8; ++j) {
      b0[j] = b0a[j]; b0[8 + j] = b0b[j];
      b1[j] = b1a[j]; b1[8 + j] = b1b[j];
    }
    acc0 = wmma16(a, b0, acc0);
    acc1 = wmma16(a, b1, acc1);
  }

  // Unified branch-free store: K -> [row][col] stride (32,1); V -> (1,64).
  const float bias0 = bias[laneLo];
  const float bias1 = bias[n1c];
  const int rs = doV ? 1 : 32;
  const int cs = doV ? 64 : 1;
  _Float16* base = (doV ? vt : kp) + (size_t)b * KV_STRIDE;
  #pragma unroll
  for (int r = 0; r < 8; ++r) {
    const int row = mtile * 16 + r + 8 * laneHi;             // 0..63 (all stored)
    base[row * rs + laneLo * cs] = (_Float16)(acc0[r] + bias0);
    base[row * rs + n1 * cs]     = (_Float16)(acc1[r] + bias1);
  }
}

// ---------------------------------------------------------------------------
// Kernel 2: attention + residual. grid = (25, 512), 8 waves/block,
// one 16-query tile per wave: dots(4 WMMA) -> softmax -> PV(4 WMMA).
// ---------------------------------------------------------------------------
__global__ __launch_bounds__(256) void attn_kernel(
    const float* __restrict__ x,
    const _Float16* __restrict__ kp, const _Float16* __restrict__ vt,
    float* __restrict__ out)
{
  __shared__ _Float16 s_attn[8][16 * 64];   // per-wave attn tile (D->A relayout)
  __shared__ float    s_x  [8][32 * 16];    // per-wave fp32 x tile (32-ch padded)
  __shared__ float    s_out[8][16 * 33];    // per-wave out tile (33-stride padded)

  const int b      = blockIdx.y;
  const int w      = threadIdx.x >> 5;
  const int lane   = threadIdx.x & 31;
  const int laneLo = lane & 15;
  const int laneHi = lane >> 4;
  const int aOff   = laneHi * 8;
  const int bOff   = laneHi * 16;

  const int T = blockIdx.x * 8 + w;
  if (T >= QTILES) return;                   // wave-uniform
  const int qbase = T * 16;

  const float* xb = x + (size_t)b * CDIM * HW;

  // --- 1. q A-frag (16 rows x 24 ch, K pad 32). Branch-free: clamp + cndmask.
  //        Also stash fp32 x tile in padded LDS for the residual add.
  v16h qa;
  #pragma unroll
  for (int j = 0; j < 8; ++j) {
    const int d0  = aOff + j;                    // always < 16 (valid)
    const int d1  = aOff + 16 + j;               // laneHi=1: 24..31 (pad)
    const int d1c = d1 < CDIM ? d1 : (CDIM - 1);
    float f0 = xb[(size_t)d0  * HW + qbase + laneLo];
    float f1 = xb[(size_t)d1c * HW + qbase + laneLo];
    float f1m = (d1 < CDIM) ? f1 : 0.f;          // K-dim pad must be zero
    qa[j]     = (_Float16)f0;
    qa[8 + j] = (_Float16)f1m;
    s_x[w][d0 * 16 + laneLo] = f0;
    s_x[w][d1 * 16 + laneLo] = f1m;              // d1<=31: padded array
  }

  // --- 2. dots = q @ k^T : 4 token tiles, fully unconditional vector loads ---
  const _Float16* kb = kp + (size_t)b * KV_STRIDE;
  float l[4][8];
  #pragma unroll
  for (int t = 0; t < 4; ++t) {
    const int tok = 16 * t + laneLo;             // 0..63, padded rows
    const v8h* kptr = (const v8h*)(kb + tok * 32 + aOff);   // 16B aligned
    const v8h k0 = kptr[0];
    const v8h k1 = kptr[2];
    v16h kf;
    #pragma unroll
    for (int j = 0; j < 8; ++j) { kf[j] = k0[j]; kf[8 + j] = k1[j]; }
    v8f acc = {};
    acc = wmma16(qa, kf, acc);
    const bool tokOK = (tok < MTOK);
    #pragma unroll
    for (int r = 0; r < 8; ++r)
      l[t][r] = tokOK ? acc[r] * SCALE : -1e30f; // v_cndmask, no branch
  }

  // --- 3. softmax per row (16 lanes x 4 tiles hold each row's 64 logits) ---
  #pragma unroll
  for (int r = 0; r < 8; ++r) {
    float m = fmaxf(fmaxf(l[0][r], l[1][r]), fmaxf(l[2][r], l[3][r]));
    #pragma unroll
    for (int s = 1; s < 16; s <<= 1) m = fmaxf(m, __shfl_xor(m, s, 16));
    float e0 = __expf(l[0][r] - m), e1 = __expf(l[1][r] - m);
    float e2 = __expf(l[2][r] - m), e3 = __expf(l[3][r] - m);
    float sum = e0 + e1 + e2 + e3;
    #pragma unroll
    for (int s = 1; s < 16; s <<= 1) sum += __shfl_xor(sum, s, 16);
    float inv = 1.0f / sum;
    l[0][r] = e0 * inv; l[1][r] = e1 * inv;
    l[2][r] = e2 * inv; l[3][r] = e3 * inv;
  }

  // --- 4. D-layout -> A-layout via per-wave LDS (same-wave DS ordering) ---
  _Float16* sa = s_attn[w];
  #pragma unroll
  for (int t = 0; t < 4; ++t)
    #pragma unroll
    for (int r = 0; r < 8; ++r)
      sa[(r + 8 * laneHi) * 64 + 16 * t + laneLo] = (_Float16)l[t][r];

  v16h pa0, pa1;
  {
    const v8h* p0a = (const v8h*)(sa + laneLo * 64 + aOff);        // 16B aligned
    const v8h* p0b = (const v8h*)(sa + laneLo * 64 + aOff + 16);
    const v8h* p1a = (const v8h*)(sa + laneLo * 64 + 32 + aOff);
    const v8h* p1b = (const v8h*)(sa + laneLo * 64 + 32 + aOff + 16);
    const v8h a0 = p0a[0], a1 = p0b[0], a2 = p1a[0], a3 = p1b[0];
    #pragma unroll
    for (int j = 0; j < 8; ++j) {
      pa0[j] = a0[j]; pa0[8 + j] = a1[j];
      pa1[j] = a2[j]; pa1[8 + j] = a3[j];
    }
  }

  // --- 5. out = attn @ v : V transposed+padded -> contiguous vector loads ---
  const _Float16* vb = vt + (size_t)b * KV_STRIDE;
  v8f o0 = {}, o1 = {};
  #pragma unroll
  for (int c = 0; c < 2; ++c) {
    const v16h pa = (c == 0) ? pa0 : pa1;
    #pragma unroll
    for (int u = 0; u < 2; ++u) {
      const int n = 16 * u + laneLo;                            // 0..31 padded
      const v8h* vptr = (const v8h*)(vb + n * 64 + 32 * c + aOff); // 16B aligned
      const v8h vv0 = vptr[0];
      const v8h vv1 = vptr[2];
      v16h vf;
      #pragma unroll
      for (int j = 0; j < 8; ++j) { vf[j] = vv0[j]; vf[8 + j] = vv1[j]; }
      if (u == 0) o0 = wmma16(pa, vf, o0);
      else        o1 = wmma16(pa, vf, o1);
    }
  }

  // --- 6. stage out tile (padded), then coalesced residual-add writeback ---
  float* so = s_out[w];
  #pragma unroll
  for (int r = 0; r < 8; ++r) {
    const int rl = r + 8 * laneHi;
    so[rl * 33 + laneLo]      = o0[r];
    so[rl * 33 + 16 + laneLo] = o1[r];       // cols 16..31, pad past 23 unused
  }
  float* ob = out + (size_t)b * CDIM * HW;
  #pragma unroll
  for (int idx = lane; idx < CDIM * 16; idx += 32) {
    const int cc = idx >> 4;
    const int rl = idx & 15;
    ob[(size_t)cc * HW + qbase + rl] = s_x[w][cc * 16 + rl] + so[rl * 33 + cc];
  }
}

// ---------------------------------------------------------------------------
extern "C" void kernel_launch(void* const* d_in, const int* in_sizes, int n_in,
                              void* d_out, int out_size, void* d_ws, size_t ws_size,
                              hipStream_t stream) {
  (void)in_sizes; (void)n_in; (void)out_size; (void)ws_size;
  const float* x  = (const float*)d_in[0];
  const float* z  = (const float*)d_in[1];
  const float* Wk = (const float*)d_in[2];
  const float* bk = (const float*)d_in[3];
  const float* Wv = (const float*)d_in[4];
  const float* bv = (const float*)d_in[5];

  _Float16* kp  = (_Float16*)d_ws;                         // 512*2048 halves
  _Float16* vtw = kp + (size_t)BATCH * KV_STRIDE;          // 512*2048 halves
  _Float16* wtw = vtw + (size_t)BATCH * KV_STRIDE;         // 2*24*768 halves

  // Kernel 0: weight transpose+convert (2*768*24 elements, 144 blocks)
  wt_kernel<<<dim3((2 * DDIM * CDIM) / 256), dim3(256), 0, stream>>>(Wk, Wv, wtw);

  kv_proj_kernel<<<dim3(BATCH), dim3(256), 0, stream>>>(z, wtw, bk, bv, kp, vtw);

  const int gx = (QTILES + 7) / 8;                         // 25
  attn_kernel<<<dim3(gx, BATCH), dim3(256), 0, stream>>>(x, kp, vtw, (float*)d_out);
}
